// binary_dense_72816875536781
// MI455X (gfx1250) — matrix operand order repaired and checked by reading, not compiled, not run on hardware
//
#include <hip/hip_runtime.h>

// ---------------------------------------------------------------------------
// Problem constants (match reference)
// ---------------------------------------------------------------------------
#define TM_     2
#define TN_     2
#define N_IN    1024
#define N_OUT   1024
#define TILE_I  512
#define TILE_O  512
#define BATCH   512
#define KDIM    (16 * N_IN)   // 16384 : expanded LUT-select contraction axis
#define KSPLIT  4             // grid-z K slices for parallelism
#define KCHUNK  (KDIM / KSPLIT)

typedef _Float16 half_t;
typedef half_t v8h  __attribute__((ext_vector_type(8)));
typedef half_t v16h __attribute__((ext_vector_type(16)));
typedef float  v8f  __attribute__((ext_vector_type(8)));

__device__ __forceinline__ float sgnf(float v) {
    return (v > 0.f) ? 1.f : ((v < 0.f) ? -1.f : 0.f);
}

// ---------------------------------------------------------------------------
// Kernel 1: build Wt[o][k*N_IN + i]  (K-contiguous / transposed weight)
//   W[k,i,o] = |gamma| * ( sign(c_all[2k,i',o'])*(1+sw)/2
//                        + sign(c_all[2k+1,i',o'])*(1-sw)/2 ) * pm[i',o']
//   with sw = sign(w1[i,o]), i'=i%512, o'=o%512
// Thread x -> i (coalesced f16 writes); reads are L2-resident (c_all 32MB).
// ---------------------------------------------------------------------------
__global__ __launch_bounds__(256) void build_wt_kernel(
    const float* __restrict__ w1,
    const float* __restrict__ c_all,
    const float* __restrict__ gamma,
    const float* __restrict__ pm,
    half_t* __restrict__ Wt) {
    const int i = blockIdx.x * blockDim.x + threadIdx.x;  // 0..1023
    const int o = blockIdx.y;                             // 0..1023
    const int it = i & (TILE_I - 1);
    const int ot = o & (TILE_O - 1);

    const float g  = fabsf(gamma[0]);
    const float wv = w1[(size_t)i * N_OUT + o];
    const float sw = sgnf(wv);
    const float ws0 = 0.5f * (1.f + sw);      // selects c_all[2k]
    const float ws1 = 0.5f * (1.f - sw);      // selects c_all[2k+1]
    const float p   = pm[it * TILE_O + ot];
    const float scale = g * p;

    const size_t cbase = (size_t)it * TILE_O + ot;
    half_t* wrow = Wt + (size_t)o * KDIM + i;

#pragma unroll
    for (int k = 0; k < 16; ++k) {
        const float c0 = c_all[(size_t)(2 * k + 0) * (TILE_I * TILE_O) + cbase];
        const float c1 = c_all[(size_t)(2 * k + 1) * (TILE_I * TILE_O) + cbase];
        const float val = scale * (sgnf(c0) * ws0 + sgnf(c1) * ws1);
        wrow[(size_t)k * N_IN] = (half_t)val;
    }
}

// ---------------------------------------------------------------------------
// Kernel 2: build A[b][k*N_IN + i] = (k == code(b,i)) ? f16(|x[b,i]|) : 0
//   code = 8*neg(x[b,i]) | 4*neg(x[b,shuf0(i)]) | 2*neg(...1) | 1*neg(...2)
//   shufm(i) = (i/512)*512 + rand_map_m[i%512]
// ---------------------------------------------------------------------------
__global__ __launch_bounds__(256) void build_a_kernel(
    const float* __restrict__ x,
    const int* __restrict__ rm0,
    const int* __restrict__ rm1,
    const int* __restrict__ rm2,
    half_t* __restrict__ A) {
    const int i = blockIdx.x * blockDim.x + threadIdx.x;  // 0..1023
    const int b = blockIdx.y;                             // 0..511
    const int t = i >> 9;          // which TM tile
    const int j = i & (TILE_I - 1);
    const int tb = t * TILE_I;

    const float* xb = x + (size_t)b * N_IN;
    const float xv = xb[i];
    const float x0 = xb[tb + rm0[j]];
    const float x1 = xb[tb + rm1[j]];
    const float x2 = xb[tb + rm2[j]];

    const int code = ((xv < 0.f) ? 8 : 0) | ((x0 < 0.f) ? 4 : 0) |
                     ((x1 < 0.f) ? 2 : 0) | ((x2 < 0.f) ? 1 : 0);
    const half_t av = (half_t)fabsf(xv);

    half_t* arow = A + (size_t)b * KDIM + i;
#pragma unroll
    for (int k = 0; k < 16; ++k)
        arow[(size_t)k * N_IN] = (k == code) ? av : (half_t)0.f;
}

// ---------------------------------------------------------------------------
// Kernel 3: part[kz][b][o] = sum_{kd in slice kz} A[b][kd] * Wt[o][kd]
// Both operands K-contiguous -> per-lane WMMA fragments are contiguous
// global_load_b128s straight from L2 (A 16MB + Wt 32MB << 192MB L2).
//
// Wave tile: 32(M) x 64(N) -> 2x4 subtiles, 8 wmma per K-step of 32.
// Block: 8 waves as 2(M) x 4(N) -> 64 x 256 block tile.
// Grid: (512/64, 1024/256, KSPLIT) = (8, 4, 4) = 128 blocks / 1024 waves.
//
// Fragment layouts (A/B in contiguous K order, C/D per CDNA5 ISA 7.12.2):
//   A 16x32 f16 : lane l(0-15) row M=l,  VGPR0-3: K=h*16..h*16+7, VGPR4-7: K=h*16+8..h*16+15
//   B 32x16 f16 : lane l(0-15) col N=l,  16 contiguous K halfs starting at h*16
//   C/D 16x16 f32: VGPR r, lane l(0-15)/h -> (M = r + 8h, N = l)
// ---------------------------------------------------------------------------
__global__ __launch_bounds__(256) void gemm_wmma_kernel(
    const half_t* __restrict__ A,    // [BATCH][KDIM]
    const half_t* __restrict__ Wt,   // [N_OUT][KDIM]
    float* __restrict__ part) {      // [KSPLIT][BATCH][N_OUT]
    const int lane = threadIdx.x & 31;
    const int wave = threadIdx.x >> 5;   // 0..7
    const int wm   = wave >> 2;          // 0..1
    const int wn   = wave & 3;           // 0..3
    const int l    = lane & 15;
    const int h    = lane >> 4;

    const int m0 = blockIdx.x * 64  + wm * 32;   // wave M base
    const int n0 = blockIdx.y * 256 + wn * 64;   // wave N base
    const int kz = blockIdx.z;                   // K slice
    const int kbeg = kz * KCHUNK;
    const int kend = kbeg + KCHUNK;

    v8f acc[2][4] = {};

    // per-lane fragment base pointers (K-contiguous rows)
    const half_t* arow[2];
#pragma unroll
    for (int mi = 0; mi < 2; ++mi)
        arow[mi] = A + (size_t)(m0 + mi * 16 + l) * KDIM + h * 16;

    const half_t* brow[4];
#pragma unroll
    for (int ni = 0; ni < 4; ++ni)
        brow[ni] = Wt + (size_t)(n0 + ni * 16 + l) * KDIM + h * 16;

    for (int k0 = kbeg; k0 < kend; k0 += 32) {
        v16h af[2];
#pragma unroll
        for (int mi = 0; mi < 2; ++mi) {
            v8h lo = *(const v8h*)(arow[mi] + k0);        // K = k0 + 16h ..
            v8h hi = *(const v8h*)(arow[mi] + k0 + 8);    // K = k0 + 16h + 8 ..
            af[mi] = __builtin_shufflevector(lo, hi, 0, 1, 2, 3, 4, 5, 6, 7,
                                             8, 9, 10, 11, 12, 13, 14, 15);
        }
        v16h bf[4];
#pragma unroll
        for (int ni = 0; ni < 4; ++ni)
            bf[ni] = *(const v16h*)(brow[ni] + k0);       // 32B contiguous K slice

#pragma unroll
        for (int mi = 0; mi < 2; ++mi)
#pragma unroll
            for (int ni = 0; ni < 4; ++ni)
                acc[mi][ni] = __builtin_amdgcn_wmma_f32_16x16x32_f16(
                    /*neg_a=*/false, af[mi], /*neg_b=*/false, bf[ni],
                    /*c_mod=*/(short)0, acc[mi][ni],
                    /*reuse_a=*/false, /*reuse_b=*/false);
    }

    // store partial D tiles (f32)
    float* pz = part + (size_t)kz * BATCH * N_OUT;
#pragma unroll
    for (int mi = 0; mi < 2; ++mi) {
#pragma unroll
        for (int ni = 0; ni < 4; ++ni) {
            const int colbase = n0 + ni * 16 + l;
#pragma unroll
            for (int r = 0; r < 8; ++r) {
                const int row = m0 + mi * 16 + h * 8 + r;
                pz[(size_t)row * N_OUT + colbase] = acc[mi][ni][r];
            }
        }
    }
}

// ---------------------------------------------------------------------------
// Kernel 4: fixed-order reduction of the KSPLIT partial planes (deterministic)
// ---------------------------------------------------------------------------
__global__ __launch_bounds__(256) void reduce_kernel(
    const float* __restrict__ part, float* __restrict__ out) {
    const int idx = (blockIdx.x * blockDim.x + threadIdx.x) * 4;
    const size_t plane = (size_t)BATCH * N_OUT;
    float4 s = *(const float4*)(part + idx);
#pragma unroll
    for (int z = 1; z < KSPLIT; ++z) {
        float4 v = *(const float4*)(part + z * plane + idx);
        s.x += v.x; s.y += v.y; s.z += v.z; s.w += v.w;
    }
    *(float4*)(out + idx) = s;
}

// ---------------------------------------------------------------------------
// Host launcher
// Inputs (setup_inputs order):
//   0:x (512,1024) f32   1:w1 (1024,1024) f32   2:c_all (32,512,512) f32
//   3:gamma () f32       4:pruning_mask (512,512) f32
//   5..7: rand_map_0/1/2 (512,) i32
// Output: (512,1024) f32
// Scratch: Wt 32MB @ 0, A 16MB @ 32MB, partials 8MB @ 48MB
// ---------------------------------------------------------------------------
extern "C" void kernel_launch(void* const* d_in, const int* in_sizes, int n_in,
                              void* d_out, int out_size, void* d_ws, size_t ws_size,
                              hipStream_t stream) {
    const float* x     = (const float*)d_in[0];
    const float* w1    = (const float*)d_in[1];
    const float* c_all = (const float*)d_in[2];
    const float* gamma = (const float*)d_in[3];
    const float* pm    = (const float*)d_in[4];
    const int*   rm0   = (const int*)d_in[5];
    const int*   rm1   = (const int*)d_in[6];
    const int*   rm2   = (const int*)d_in[7];

    char* ws = (char*)d_ws;
    half_t* Wt  = (half_t*)ws;                                        // 32 MB
    half_t* A   = (half_t*)(ws + (size_t)N_OUT * KDIM * sizeof(half_t)); // +16 MB
    float*  prt = (float*)(ws + (size_t)N_OUT * KDIM * sizeof(half_t)
                              + (size_t)BATCH * KDIM * sizeof(half_t)); // +8 MB

    build_wt_kernel<<<dim3(N_IN / 256, N_OUT), 256, 0, stream>>>(w1, c_all, gamma, pm, Wt);
    build_a_kernel<<<dim3(N_IN / 256, BATCH), 256, 0, stream>>>(x, rm0, rm1, rm2, A);
    gemm_wmma_kernel<<<dim3(BATCH / 64, N_OUT / 256, KSPLIT), 256, 0, stream>>>(A, Wt, prt);
    reduce_kernel<<<(BATCH * N_OUT) / (4 * 256), 256, 0, stream>>>(prt, (float*)d_out);
}
